// MultiHeadAttention_55894704390646
// MI455X (gfx1250) — matrix operand order repaired and checked by reading, not compiled, not run on hardware
//
#include <hip/hip_runtime.h>

// ---------------- problem constants ----------------
#define Bn 2
#define Sn 2048
#define En 1024
#define Hn 16
#define Dn 64
#define Mn (Bn * Sn)   // 4096 rows

typedef __attribute__((ext_vector_type(16))) __bf16 v16bf;
typedef __attribute__((ext_vector_type(8)))  __bf16 bf16x8;
typedef __attribute__((ext_vector_type(8)))  float  v8f;

// ---------------- WMMA wrapper ----------------
__device__ __forceinline__ v8f wmma_bf16(v16bf a, v16bf b, v8f c) {
    return __builtin_amdgcn_wmma_f32_16x16x32_bf16(
        false, a, false, b, (short)0, c, false, false);
}

// A operand (16x32 bf16): lane m=lane&15 -> row; element e -> col0 + 16*(e>>3) + 8*ha + (e&7)
__device__ __forceinline__ v16bf load_A(const __bf16* base, int row0, int ld, int col0, int lane) {
    int m  = lane & 15;
    int ha = lane >> 4;
    const __bf16* p = base + (row0 + m) * ld + col0 + 8 * ha;
    bf16x8 lo = *(const bf16x8*)(p);
    bf16x8 hi = *(const bf16x8*)(p + 16);
    v16bf a;
#pragma unroll
    for (int i = 0; i < 8; ++i) { a[i] = lo[i]; a[i + 8] = hi[i]; }
    return a;
}

// B operand (32x16 bf16): lane n=lane&15 -> src row nrow0+n ; element e -> k0 + 16*(e>>3) + 8*hb + (e&7)
__device__ __forceinline__ v16bf load_B(const __bf16* base, int nrow0, int ld, int k0, int lane) {
    int n  = lane & 15;
    int hb = lane >> 4;
    const __bf16* p = base + (nrow0 + n) * ld + k0 + 8 * hb;
    bf16x8 lo = *(const bf16x8*)(p);
    bf16x8 hi = *(const bf16x8*)(p + 16);
    v16bf b;
#pragma unroll
    for (int i = 0; i < 8; ++i) { b[i] = lo[i]; b[i + 8] = hi[i]; }
    return b;
}

// ---------------- f32 -> bf16 conversion ----------------
__global__ __launch_bounds__(256) void cvt_bf16(const float* __restrict__ s,
                                                __bf16* __restrict__ d, int n4) {
    int i = (blockIdx.x * 256 + threadIdx.x) * 4;
    if (i + 3 < n4 * 4) {
        float4 v = *(const float4*)(s + i);
        d[i + 0] = (__bf16)v.x;
        d[i + 1] = (__bf16)v.y;
        d[i + 2] = (__bf16)v.z;
        d[i + 3] = (__bf16)v.w;
    }
}

// ---------------- GEMM operand set + helpers (32M x 64N wave tile) ----------------
struct GemmOps { v16bf a0, a1, b[4]; };

__device__ __forceinline__ void gemm_load(GemmOps& o, const __bf16* X, const __bf16* W,
                                          int m0, int n0, int k, int lane) {
    o.a0 = load_A(X, m0,      En, k, lane);
    o.a1 = load_A(X, m0 + 16, En, k, lane);
#pragma unroll
    for (int t = 0; t < 4; ++t) o.b[t] = load_B(W, n0 + t * 16, En, k, lane);
}

__device__ __forceinline__ void gemm_mma(v8f (&acc)[2][4], const GemmOps& o) {
#pragma unroll
    for (int t = 0; t < 4; ++t) {
        acc[0][t] = wmma_bf16(o.a0, o.b[t], acc[0][t]);
        acc[1][t] = wmma_bf16(o.a1, o.b[t], acc[1][t]);
    }
}

// Fully unrolled K loop (32 steps): straight-line code, no loop-carried register
// constraints -> no accumulator/operand rotation copies; scheduler hoists loads.
__device__ __forceinline__ void gemm_core(v8f (&acc)[2][4], const __bf16* X, const __bf16* W,
                                          int m0, int n0, int lane) {
#pragma unroll
    for (int k0 = 0; k0 < En; k0 += 32) {
        GemmOps p;
        gemm_load(p, X, W, m0, n0, k0, lane);
        gemm_mma(acc, p);
    }
}

// ---------------- QKV projection: out = x @ W^T ----------------
// z=0 -> Q (scaled 1/8) [b,h,s,d] ; z=1 -> K [b,h,s,d] ; z=2 -> V transposed [b,h,d,s]
__global__ __launch_bounds__(256) void gemm_qkv(const __bf16* __restrict__ xb,
                                                const __bf16* __restrict__ wq,
                                                const __bf16* __restrict__ wk,
                                                const __bf16* __restrict__ wv,
                                                __bf16* __restrict__ Qb,
                                                __bf16* __restrict__ Kb,
                                                __bf16* __restrict__ Vt) {
    int lane = threadIdx.x & 31;
    int wave = threadIdx.x >> 5;
    int m0 = (blockIdx.x * 8 + wave) * 32;
    int n0 = blockIdx.y * 64;
    int z  = blockIdx.z;
    const __bf16* W = (z == 0) ? wq : ((z == 1) ? wk : wv);

    v8f acc[2][4];
#pragma unroll
    for (int i = 0; i < 2; ++i)
#pragma unroll
        for (int t = 0; t < 4; ++t) acc[i][t] = v8f{};

    gemm_core(acc, xb, W, m0, n0, lane);

    float scale = (z == 0) ? 0.125f : 1.0f;   // fold 1/sqrt(D) into Q
    int h = lane >> 4, nl = lane & 15;
#pragma unroll
    for (int mi = 0; mi < 2; ++mi) {
#pragma unroll
        for (int t = 0; t < 4; ++t) {
#pragma unroll
            for (int r = 0; r < 8; ++r) {
                int mrow = m0 + mi * 16 + r + 8 * h;   // = b*S + s
                int bb = mrow >> 11;
                int s  = mrow & (Sn - 1);
                int n  = n0 + t * 16 + nl;             // = head*64 + d
                int head = n >> 6;
                int d    = n & 63;
                __bf16 v = (__bf16)(acc[mi][t][r] * scale);
                if (z == 2)
                    Vt[((bb * Hn + head) * Dn + d) * Sn + s] = v;   // [b,h,d,s]
                else if (z == 0)
                    Qb[((bb * Hn + head) * Sn + s) * Dn + d] = v;   // [b,h,s,d]
                else
                    Kb[((bb * Hn + head) * Sn + s) * Dn + d] = v;
            }
        }
    }
}

// ---------------- flash attention ----------------
__device__ __forceinline__ void loadK(v16bf (&ak)[4], const __bf16* Kp, int k0, int lane) {
#pragma unroll
    for (int t = 0; t < 2; ++t)
#pragma unroll
        for (int w = 0; w < 2; ++w)
            ak[t * 2 + w] = load_A(Kp, k0 + 16 * t, Dn, w * 32, lane);
}

__device__ __forceinline__ void loadV(v16bf (&bv)[4], const __bf16* Vp, int k0, int lane) {
#pragma unroll
    for (int dc = 0; dc < 4; ++dc) bv[dc] = load_B(Vp, dc * 16, Sn, k0, lane);
}

// ak[t*2+w]: K rows k0+16t, d-range w*32 ; bv[dc]: V^T rows dc*16, keys k0+16*(e>>3)+8hb+(e&7)
__device__ __forceinline__ void attn_chunk(int k0, int lane, int q0,
                                           const v16bf (&ak)[4], const v16bf (&bv)[4],
                                           const v16bf (&Bq)[2][2],
                                           v8f (&ctxacc)[2][4],
                                           float (&mrun)[2], float (&lrun)[2]) {
    int h = lane >> 4, nl = lane & 15;
#pragma unroll
    for (int sub = 0; sub < 2; ++sub) {
        v8f c0 = v8f{}, c1 = v8f{};
#pragma unroll
        for (int w = 0; w < 2; ++w) {
            c0 = wmma_bf16(ak[0 * 2 + w], Bq[sub][w], c0);
            c1 = wmma_bf16(ak[1 * 2 + w], Bq[sub][w], c1);
        }
        int q_lane = q0 + sub * 16 + nl;
        float mloc = -1e30f;
#pragma unroll
        for (int r = 0; r < 8; ++r) {
            int key0 = k0 + r + 8 * h;
            int key1 = key0 + 16;
            float v0 = (key0 <= q_lane) ? c0[r] : -1e30f;
            float v1 = (key1 <= q_lane) ? c1[r] : -1e30f;
            c0[r] = v0; c1[r] = v1;
            mloc = fmaxf(mloc, fmaxf(v0, v1));
        }
        mloc = fmaxf(mloc, __shfl_xor(mloc, 16, 32));
        float mnew = fmaxf(mrun[sub], mloc);
        float f = __expf(mrun[sub] - mnew);

        float ladd = 0.0f;
        v16bf ap;      // S^T accumulator layout == A-operand layout of P
#pragma unroll
        for (int r = 0; r < 8; ++r) {
            float p0 = __expf(c0[r] - mnew);
            float p1 = __expf(c1[r] - mnew);
            ladd += p0 + p1;
            ap[r]     = (__bf16)p0;
            ap[r + 8] = (__bf16)p1;
        }
        ladd += __shfl_xor(ladd, 16, 32);
        lrun[sub] = lrun[sub] * f + ladd;
        mrun[sub] = mnew;

        float fq[8];
#pragma unroll
        for (int r = 0; r < 8; ++r) fq[r] = __shfl(f, r + 8 * h, 32);
#pragma unroll
        for (int dc = 0; dc < 4; ++dc) {
#pragma unroll
            for (int r = 0; r < 8; ++r) ctxacc[sub][dc][r] *= fq[r];
            ctxacc[sub][dc] = wmma_bf16(ap, bv[dc], ctxacc[sub][dc]);
        }
    }
}

// One wave per 32-row q tile (two 16-row subtiles sharing K/V loads).
// Ping-pong prefetch of K operands across the chunk loop.
__global__ __launch_bounds__(256) void attn_kernel(const __bf16* __restrict__ Qb,
                                                   const __bf16* __restrict__ Kb,
                                                   const __bf16* __restrict__ Vt,
                                                   __bf16* __restrict__ ctx) {
    int lane = threadIdx.x & 31;
    int wave = threadIdx.x >> 5;
    int gw = blockIdx.x * 8 + wave;        // 0 .. 2047
    int bh = gw >> 6;                      // / (S/32)
    int qt = gw & 63;
    int b = bh >> 4, head = bh & 15;
    int q0 = qt * 32;

    const __bf16* Qp = Qb + (size_t)(b * Hn + head) * Sn * Dn;   // [s][d]
    const __bf16* Kp = Kb + (size_t)(b * Hn + head) * Sn * Dn;   // [s][d]
    const __bf16* Vp = Vt + (size_t)(b * Hn + head) * Dn * Sn;   // [d][s]

    int h = lane >> 4, nl = lane & 15;

    v16bf Bq[2][2];
#pragma unroll
    for (int sub = 0; sub < 2; ++sub)
#pragma unroll
        for (int w = 0; w < 2; ++w)
            Bq[sub][w] = load_B(Qp, q0 + sub * 16, Dn, w * 32, lane);

    v8f ctxacc[2][4];
#pragma unroll
    for (int i = 0; i < 2; ++i)
#pragma unroll
        for (int t = 0; t < 4; ++t) ctxacc[i][t] = v8f{};
    float mrun[2] = {-1e30f, -1e30f};
    float lrun[2] = {0.0f, 0.0f};

    int kend = q0;                         // last chunk base (q0 multiple of 32)

    v16bf akA[4], akB[4];
    loadK(akA, Kp, 0, lane);

    int k0 = 0;
    for (;;) {
        {   // consume akA, prefetch akB
            if (k0 < kend) loadK(akB, Kp, k0 + 32, lane);
            v16bf bv[4];
            loadV(bv, Vp, k0, lane);
            attn_chunk(k0, lane, q0, akA, bv, Bq, ctxacc, mrun, lrun);
            if (k0 >= kend) break;
            k0 += 32;
        }
        {   // consume akB, prefetch akA
            if (k0 < kend) loadK(akA, Kp, k0 + 32, lane);
            v16bf bv[4];
            loadV(bv, Vp, k0, lane);
            attn_chunk(k0, lane, q0, akB, bv, Bq, ctxacc, mrun, lrun);
            if (k0 >= kend) break;
            k0 += 32;
        }
    }

    // epilogue: divide by l, store ctx bf16 as [b,s,h,d]
#pragma unroll
    for (int sub = 0; sub < 2; ++sub) {
        float lq[8];
#pragma unroll
        for (int r = 0; r < 8; ++r) lq[r] = __shfl(lrun[sub], r + 8 * h, 32);
#pragma unroll
        for (int dc = 0; dc < 4; ++dc) {
#pragma unroll
            for (int r = 0; r < 8; ++r) {
                int q = q0 + sub * 16 + r + 8 * h;
                int d = dc * 16 + nl;
                float v = ctxacc[sub][dc][r] / lq[r];
                ctx[((size_t)(b * Sn + q) * Hn + head) * Dn + d] = (__bf16)v;
            }
        }
    }
}

// ---------------- output projection: out = ctx @ Wo^T + bo (f32 out) ----------------
__global__ __launch_bounds__(256) void gemm_out(const __bf16* __restrict__ ctxb,
                                                const __bf16* __restrict__ wob,
                                                const float* __restrict__ bo,
                                                float* __restrict__ out) {
    int lane = threadIdx.x & 31;
    int wave = threadIdx.x >> 5;
    int m0 = (blockIdx.x * 8 + wave) * 32;
    int n0 = blockIdx.y * 64;

    v8f acc[2][4];
#pragma unroll
    for (int i = 0; i < 2; ++i)
#pragma unroll
        for (int t = 0; t < 4; ++t) acc[i][t] = v8f{};

    gemm_core(acc, ctxb, wob, m0, n0, lane);

    int h = lane >> 4, nl = lane & 15;
#pragma unroll
    for (int mi = 0; mi < 2; ++mi) {
#pragma unroll
        for (int t = 0; t < 4; ++t) {
            int n = n0 + t * 16 + nl;
            float bias = bo[n];
#pragma unroll
            for (int r = 0; r < 8; ++r) {
                int mrow = m0 + mi * 16 + r + 8 * h;
                out[(size_t)mrow * En + n] = acc[mi][t][r] + bias;
            }
        }
    }
}

// ---------------- host ----------------
extern "C" void kernel_launch(void* const* d_in, const int* in_sizes, int n_in,
                              void* d_out, int out_size, void* d_ws, size_t ws_size,
                              hipStream_t stream) {
    const float* x  = (const float*)d_in[0];
    const float* Wq = (const float*)d_in[1];
    const float* Wk = (const float*)d_in[2];
    const float* Wv = (const float*)d_in[3];
    const float* Wo = (const float*)d_in[4];
    const float* bo = (const float*)d_in[5];
    float* out = (float*)d_out;

    __bf16* ws = (__bf16*)d_ws;
    const size_t xe = (size_t)Mn * En;
    const size_t we = (size_t)En * En;
    const size_t qe = (size_t)Bn * Hn * Sn * Dn;
    __bf16* xb  = ws;            size_t off = xe;
    __bf16* wqb = ws + off;      off += we;
    __bf16* wkb = ws + off;      off += we;
    __bf16* wvb = ws + off;      off += we;
    __bf16* wob = ws + off;      off += we;
    __bf16* Qb  = ws + off;      off += qe;
    __bf16* Kb  = ws + off;      off += qe;
    __bf16* Vt  = ws + off;      off += qe;
    __bf16* ctx = ws + off;      off += xe;
    (void)ws_size; (void)in_sizes; (void)n_in; (void)out_size;

    cvt_bf16<<<(int)(xe / 4 / 256), 256, 0, stream>>>(x,  xb,  (int)(xe / 4));
    cvt_bf16<<<(int)(we / 4 / 256), 256, 0, stream>>>(Wq, wqb, (int)(we / 4));
    cvt_bf16<<<(int)(we / 4 / 256), 256, 0, stream>>>(Wk, wkb, (int)(we / 4));
    cvt_bf16<<<(int)(we / 4 / 256), 256, 0, stream>>>(Wv, wvb, (int)(we / 4));
    cvt_bf16<<<(int)(we / 4 / 256), 256, 0, stream>>>(Wo, wob, (int)(we / 4));

    gemm_qkv<<<dim3(Mn / 256, En / 64, 3), 256, 0, stream>>>(xb, wqb, wkb, wvb, Qb, Kb, Vt);

    attn_kernel<<<256, 256, 0, stream>>>(Qb, Kb, Vt, ctx);

    gemm_out<<<dim3(Mn / 256, En / 64, 1), 256, 0, stream>>>(ctx, wob, bo, out);
}
